// ColorAttentionModule_41815801594256
// MI455X (gfx1250) — matrix operand order repaired and checked by reading, not compiled, not run on hardware
//
#include <hip/hip_runtime.h>
#include <hip/hip_bf16.h>
#include <math.h>

typedef __attribute__((ext_vector_type(16))) _Float16 v16h;
typedef __attribute__((ext_vector_type(8)))  float    v8f;

#define HW   65536   // 256*256
#define NC   64
#define TP   128     // pixels per workgroup tile
#define BNEPS 1e-5f

// ---------------------------------------------------------------------------
// Kernel 1: per-(b,c) 256-bin histogram + argmax (ties -> lowest bin)
// ---------------------------------------------------------------------------
__global__ __launch_bounds__(256) void hist_kernel(const float* __restrict__ x,
                                                   int* __restrict__ dom) {
    __shared__ unsigned hist[256];
    __shared__ unsigned keys[256];
    const int t   = threadIdx.x;
    const int row = blockIdx.x;            // b*64 + c, 1024 rows
    hist[t] = 0u;
    __syncthreads();

    const float4* xr = reinterpret_cast<const float4*>(x + (size_t)row * HW);
    for (int j = 0; j < 64; ++j) {         // 64 * 256 * 4 = 65536 px
        const float4 v = xr[t + 256 * j];
        const float vv[4] = {v.x, v.y, v.z, v.w};
        #pragma unroll
        for (int e = 0; e < 4; ++e) {
            const float f = vv[e];
            if (f >= 0.0f && f <= 255.0f) {            // histc ignores OOR
                int bin = (int)(f * (256.0f / 255.0f)); // trunc == floor (f>=0)
                bin = bin > 255 ? 255 : bin;
                atomicAdd(&hist[bin], 1u);
            }
        }
    }
    __syncthreads();

    // key = (count << 8) | (255 - bin): max key == max count, tie -> lower bin
    keys[t] = (hist[t] << 8) | (unsigned)(255 - t);
    __syncthreads();
    for (int s = 128; s > 0; s >>= 1) {
        if (t < s) {
            unsigned a = keys[t], b = keys[t + s];
            keys[t] = a > b ? a : b;
        }
        __syncthreads();
    }
    if (t == 0) dom[row] = 255 - (int)(keys[0] & 255u);
}

// ---------------------------------------------------------------------------
// Kernel 2: fused mask -> WMMA conv1 -> bn1/relu -> conv2 -> bn2/sigmoid -> x*s
// One workgroup = 128 contiguous pixels x 64 channels. 8 waves, each wave
// owns a 16-pixel subtile and does the [16x64]x[64x64] GEMM with 8 WMMAs.
// ---------------------------------------------------------------------------
__global__ __launch_bounds__(256) void fused_kernel(
    const float* __restrict__ x,
    const float* __restrict__ w1,  const float* __restrict__ b1,
    const float* __restrict__ g1,  const float* __restrict__ be1,
    const float* __restrict__ mu1, const float* __restrict__ va1,
    const float* __restrict__ w2,  const float* __restrict__ b2,
    const float* __restrict__ g2,  const float* __restrict__ be2,
    const float* __restrict__ mu2, const float* __restrict__ va2,
    const int*   __restrict__ dom,
    float* __restrict__ out) {

    __shared__ float     xs[NC * TP];       // 32 KB: x tile
    __shared__ _Float16  ms[NC * TP];       // 16 KB: binary mask tile (f16)
    __shared__ _Float16  whB[NC * 68];      // ~8.5 KB: W1^T in f16, padded stride
    __shared__ float     siglds[TP];        // per-pixel attention scalar

    const int t   = threadIdx.x;
    const int blk = blockIdx.x;             // 8192 = 16 batches * 512 tiles
    const int b   = blk >> 9;
    const int p0  = (blk & 511) * TP;
    const size_t xbase = (size_t)b * NC * HW;

    // Stage conv1_w transposed: whB[k][o] = w1[o][k], f16, stride 68 halves.
    #pragma unroll
    for (int j = 0; j < 16; ++j) {
        const int idx = t + 256 * j;        // 4096 elements
        const int o = idx & 63, k = idx >> 6;
        whB[k * 68 + o] = (_Float16)w1[o * 64 + k];
    }

    // Stage x and mask tiles (float4 loads; 8 chunks/thread).
    #pragma unroll
    for (int j = 0; j < 8; ++j) {
        const int q = t + 256 * j;          // 2048 float4 chunks
        const int c = q >> 5;               // 32 chunks per channel row
        const int p = (q & 31) * 4;
        const float4 v = *reinterpret_cast<const float4*>(
            x + xbase + (size_t)c * HW + p0 + p);
        *reinterpret_cast<float4*>(&xs[c * TP + p]) = v;
        const int dm = dom[b * NC + c];
        ms[c * TP + p + 0] = (_Float16)(float)((int)floorf(v.x) != dm);
        ms[c * TP + p + 1] = (_Float16)(float)((int)floorf(v.y) != dm);
        ms[c * TP + p + 2] = (_Float16)(float)((int)floorf(v.z) != dm);
        ms[c * TP + p + 3] = (_Float16)(float)((int)floorf(v.w) != dm);
    }
    __syncthreads();

    const int lane = t & 31;
    const int wv   = t >> 5;
    const int ps   = wv * 16;               // wave's pixel subtile base
    const int rowA = lane & 15;
    const int hsel = (lane >> 4) * 8;       // K-half select per ISA A layout

    // A registers: 16x32 f16 per K-chunk. Element i of lane L holds
    // A[row = L&15][k = kbase + i + (i>>3)*8 + (L>>4)*8].
    v16h avec[2];
    #pragma unroll
    for (int kc = 0; kc < 2; ++kc) {
        const int kbase = kc * 32;
        #pragma unroll
        for (int i = 0; i < 16; ++i) {
            const int kk = kbase + hsel + i + ((i >> 3) << 3);
            avec[kc][i] = ms[kk * TP + ps + rowA];
        }
    }

    // GEMM: D[16x16] tiles over 4 N-chunks, K=64 in two 32-steps.
    v8f acc[4];
    #pragma unroll
    for (int n = 0; n < 4; ++n) {
        v8f c = {};
        #pragma unroll
        for (int kc = 0; kc < 2; ++kc) {
            const int kbase = kc * 32;
            v16h bvec;                       // B[k = kbase + i + (i>>3)*8 + (L>>4)*8][n = nbase + (L&15)]
            #pragma unroll
            for (int i = 0; i < 16; ++i)
                bvec[i] = whB[(kbase + hsel + i + ((i >> 3) << 3)) * 68 + n * 16 + rowA];
            c = __builtin_amdgcn_wmma_f32_16x16x32_f16(
                    false, avec[kc], false, bvec, (short)0, c, false, false);
        }
        acc[n] = c;
    }

    // Epilogue: (dot + b1)*bn1 -> relu -> * w2, accumulate per pixel.
    // D layout: lane holds col o = nbase + (lane&15); rows M = r + hsel.
    float wsum[8];
    #pragma unroll
    for (int r = 0; r < 8; ++r) wsum[r] = 0.0f;
    const int ncol = lane & 15;
    #pragma unroll
    for (int n = 0; n < 4; ++n) {
        const int o = n * 16 + ncol;
        const float inv = g1[o] * rsqrtf(va1[o] + BNEPS);
        const float tt  = be1[o] + (b1[o] - mu1[o]) * inv;
        const float wo  = w2[o];
        #pragma unroll
        for (int r = 0; r < 8; ++r) {
            float y = fmaxf(acc[n][r] * inv + tt, 0.0f);
            wsum[r] += y * wo;
        }
    }
    // Sum the 16 o-columns held across each 16-lane half (xor masks stay in-half).
    #pragma unroll
    for (int m = 1; m < 16; m <<= 1) {
        #pragma unroll
        for (int r = 0; r < 8; ++r)
            wsum[r] += __shfl_xor(wsum[r], m, 32);
    }
    if ((lane & 15) == 0) {
        const float inv2 = g2[0] * rsqrtf(va2[0] + BNEPS);
        const float t2   = be2[0] + (b2[0] - mu2[0]) * inv2;
        #pragma unroll
        for (int r = 0; r < 8; ++r) {
            const float z = wsum[r] * inv2 + t2;
            siglds[ps + hsel + r] = 1.0f / (1.0f + __expf(-z));
        }
    }
    __syncthreads();

    // out = x * sigmoid, x re-read from LDS (HBM read of x happens once here).
    #pragma unroll
    for (int j = 0; j < 8; ++j) {
        const int q = t + 256 * j;
        const int c = q >> 5;
        const int p = (q & 31) * 4;
        const float4 v = *reinterpret_cast<const float4*>(&xs[c * TP + p]);
        float4 o4;
        o4.x = v.x * siglds[p + 0];
        o4.y = v.y * siglds[p + 1];
        o4.z = v.z * siglds[p + 2];
        o4.w = v.w * siglds[p + 3];
        *reinterpret_cast<float4*>(out + xbase + (size_t)c * HW + p0 + p) = o4;
    }
}

// ---------------------------------------------------------------------------
extern "C" void kernel_launch(void* const* d_in, const int* in_sizes, int n_in,
                              void* d_out, int out_size, void* d_ws, size_t ws_size,
                              hipStream_t stream) {
    const float* x   = (const float*)d_in[0];
    const float* w1  = (const float*)d_in[1];
    const float* b1  = (const float*)d_in[2];
    const float* g1  = (const float*)d_in[3];
    const float* be1 = (const float*)d_in[4];
    const float* mu1 = (const float*)d_in[5];
    const float* va1 = (const float*)d_in[6];
    const float* w2  = (const float*)d_in[7];
    const float* b2  = (const float*)d_in[8];
    const float* g2  = (const float*)d_in[9];
    const float* be2 = (const float*)d_in[10];
    const float* mu2 = (const float*)d_in[11];
    const float* va2 = (const float*)d_in[12];
    float* out = (float*)d_out;
    int* dom = (int*)d_ws;                  // 1024 ints of scratch

    hist_kernel<<<dim3(16 * NC), dim3(256), 0, stream>>>(x, dom);
    fused_kernel<<<dim3(16 * (HW / TP)), dim3(256), 0, stream>>>(
        x, w1, b1, g1, be1, mu1, va1, w2, b2, g2, be2, mu2, va2, dom, out);
}
